// Banyan_13864154431809
// MI455X (gfx1250) — compile-verified, matrix operand order checked by reading, not run on hardware
//
#include <hip/hip_runtime.h>
#include <hip/hip_bf16.h>
#include <math.h>
#include <stdint.h>

typedef __attribute__((ext_vector_type(16))) _Float16 v16h;
typedef __attribute__((ext_vector_type(8)))  float    v8f;
typedef __attribute__((ext_vector_type(4)))  unsigned int u32x4;
typedef __attribute__((ext_vector_type(8)))  int      i32x8;
typedef __attribute__((ext_vector_type(4)))  int      i32x4;

#define B_  48
#define S_  160
#define V_  32000
#define E_  16
#define NODE_CAP 16384
#define NTCAP (NODE_CAP/16)              /* 1024 node tiles */
#define MERGES_PER_ROW (S_-1)            /* 159 */
#define TCAP (B_*MERGES_PER_ROW*2)       /* 15264 */
#define TILES (TCAP/16)                  /* 954  */

/* ---- workspace layout, in 32-bit units ---- */
#define WS_CNT   0                       /* [0]=Ntot [1]=Tvalid [2]=Lcount */
#define WS_FLAGS 16
#define WS_LEAF  (WS_FLAGS + 32768)
#define WS_NC    (WS_LEAF  + 32768)      /* float[NODE_CAP*E_] */
#define WS_TRAW  (WS_NC    + NODE_CAP*E_)
#define WS_TCMP  (WS_TRAW  + TCAP*4)
#define WS_LOSS  (WS_TCMP  + TCAP*4)     /* float[TCAP] per-row loss */
#define WS_BPACK (WS_LOSS  + TCAP)       /* f16 WMMA-ready B panel: NTCAP*32*16 halves */
#define WS_TOTAL (WS_BPACK + NTCAP*32*16/2)

#if __has_builtin(__builtin_amdgcn_tensor_load_to_lds) && \
    __has_builtin(__builtin_amdgcn_s_wait_tensorcnt)
#define HAVE_TDM 1
#else
#define HAVE_TDM 0
#endif

__global__ void k_zero(int* wsI) {
  for (int i = blockIdx.x * blockDim.x + threadIdx.x; i < WS_TOTAL;
       i += gridDim.x * blockDim.x)
    wsI[i] = 0;
}

__global__ void k_mark(const int* __restrict__ seqs, int* wsI) {
  int i = blockIdx.x * blockDim.x + threadIdx.x;
  if (i >= B_ * S_) return;
  int tok = seqs[i];
  if (tok >= 0 && tok < V_ - 1) wsI[WS_FLAGS + tok] = 1;   /* pad token V-1 excluded */
}

/* single-block ordered scan over the vocab flag array -> sorted-unique leaf ids,
   gather leaf embedding rows into the node pool. */
__global__ __launch_bounds__(1024)
void k_scan(const float* __restrict__ emb, int* wsI, float* wsF) {
  __shared__ int part[1024];
  const int t = threadIdx.x;
  const int CHUNK = 32;                         /* 1024*32 >= 32000 */
  int begin = t * CHUNK;
  int end   = begin + CHUNK; if (end > V_) end = V_;
  int* flags  = wsI + WS_FLAGS;
  int* leafid = wsI + WS_LEAF;
  float* n_c  = wsF + WS_NC;

  int c = 0;
  if (begin < V_) for (int i = begin; i < end; ++i) c += flags[i];
  part[t] = c;
  __syncthreads();
  for (int off = 1; off < 1024; off <<= 1) {
    int v = part[t];
    int add = (t >= off) ? part[t - off] : 0;
    __syncthreads();
    part[t] = v + add;
    __syncthreads();
  }
  int id = part[t] - c;                          /* exclusive prefix */
  if (begin < V_) {
    for (int i = begin; i < end; ++i) {
      if (flags[i]) {
        leafid[i] = id;
        for (int e = 0; e < E_; ++e) n_c[(size_t)id * E_ + e] = emb[(size_t)i * E_ + e];
        ++id;
      } else {
        leafid[i] = -1;
      }
    }
  }
  if (t == 1023) {
    int L = part[1023];
    wsI[2] = L;                                  /* leaf count */
    wsI[0] = L + B_ * MERGES_PER_ROW;            /* total node count */
  }
}

/* one wave32 per batch row: sequential greedy cosine merge, records targets */
__global__ __launch_bounds__(32)
void k_build(const int* __restrict__ seqs,
             const float* __restrict__ comp_l, const float* __restrict__ comp_r,
             const float* __restrict__ cbp, int* wsI, float* wsF) {
  __shared__ int   idx[S_];
  __shared__ float nrm[S_];
  __shared__ int   shW;
  const int b = blockIdx.x;
  const int lane = threadIdx.x;
  const int* leafid = wsI + WS_LEAF;
  float* n_c = wsF + WS_NC;
  int*   traw = wsI + WS_TRAW;
  const int Lc = wsI[2];

  float sl[4], sr[4], cbv[4];
  for (int i = 0; i < 4; ++i) {
    sl[i] = 1.f / (1.f + __expf(-comp_l[i]));
    sr[i] = 1.f / (1.f + __expf(-comp_r[i]));
    cbv[i] = cbp[i];
  }

  if (lane == 0) {
    int w = 0;
    for (int i = 0; i < S_; ++i) {
      int tok = seqs[b * S_ + i];
      if (tok != V_ - 1) {
        int id = leafid[tok];
        if (id >= 0) idx[w++] = id;
      }
    }
    shW = w;
  }
  __syncthreads();
  int W = shW;
  for (int i = lane; i < W; i += 32) {
    const float* v = n_c + (size_t)idx[i] * E_;
    float s = 0.f;
    for (int e = 0; e < E_; ++e) s += v[e] * v[e];
    nrm[i] = fmaxf(sqrtf(s), 1e-8f);
  }
  __syncthreads();

  int step = 0;
  while (W > 1) {
    float bv = -INFINITY; int bi = 0x7fffffff;
    for (int p = lane; p < W - 1; p += 32) {
      const float* lv = n_c + (size_t)idx[p] * E_;
      const float* rv = n_c + (size_t)idx[p + 1] * E_;
      float d = 0.f;
      for (int e = 0; e < E_; ++e) d += lv[e] * rv[e];
      float cs = d / (nrm[p] * nrm[p + 1]);
      if (cs > bv || (cs == bv && p < bi)) { bv = cs; bi = p; }
    }
    for (int off = 16; off >= 1; off >>= 1) {
      float ov = __shfl_xor(bv, off, 32);
      int   oi = __shfl_xor(bi, off, 32);
      if (ov > bv || (ov == bv && oi < bi)) { bv = ov; bi = oi; }
    }
    const int p = bi;
    const int dst = Lc + b * MERGES_PER_ROW + step;
    if (lane < E_) {
      float lval = n_c[(size_t)idx[p] * E_ + lane];
      float rval = n_c[(size_t)idx[p + 1] * E_ + lane];
      n_c[(size_t)dst * E_ + lane] =
          sl[lane & 3] * lval + sr[lane & 3] * rval + cbv[lane & 3];
    }
    if (lane == 0) {
      int slot = (b * MERGES_PER_ROW + step) * 2;
      int rn = (p + 2 <= W - 1) ? p + 2 : -1;
      traw[slot * 4 + 0] = p;     traw[slot * 4 + 1] = p - 1;
      traw[slot * 4 + 2] = p + 1; traw[slot * 4 + 3] = (p - 1 >= 0) ? 1 : 0;
      traw[(slot + 1) * 4 + 0] = p + 1; traw[(slot + 1) * 4 + 1] = p;
      traw[(slot + 1) * 4 + 2] = rn;    traw[(slot + 1) * 4 + 3] = (rn >= 0) ? 1 : 0;
      for (int i = p; i < W - 1; ++i) { idx[i] = idx[i + 1]; nrm[i] = nrm[i + 1]; }
    }
    __syncthreads();
    --W; ++step;
    __syncthreads();
  }
}

/* ordered compaction of valid target triples (deterministic) */
__global__ void k_compact(int* wsI) {
  if (threadIdx.x != 0 || blockIdx.x != 0) return;
  const int* raw = wsI + WS_TRAW;
  int* cmp = wsI + WS_TCMP;
  int T = 0;
  for (int i = 0; i < TCAP; ++i) {
    if (raw[i * 4 + 3]) {
      cmp[T * 4 + 0] = raw[i * 4 + 0];
      cmp[T * 4 + 1] = raw[i * 4 + 1];
      cmp[T * 4 + 2] = raw[i * 4 + 2];
      cmp[T * 4 + 3] = 1;
      ++T;
    }
  }
  wsI[1] = T;
}

/* Pre-bake the node pool into a WMMA-lane-layout f16 B panel:
   tile j, lane l -> 16 halves at bp[(j*32+l)*16]; halves 0..7 = n_c[j*16+(l&15)]
   [e0..e0+7] (e0 = 0 for lanes<16, 8 otherwise), halves 8..15 = 0 (K pad 16->32). */
__global__ void k_pack(int* wsI, float* wsF) {
  int gid = blockIdx.x * blockDim.x + threadIdx.x;
  if (gid >= NTCAP * 32) return;
  const int tile = gid >> 5, l = gid & 31;
  const int Ntot = wsI[0];
  const int node = tile * 16 + (l & 15);
  const int e0 = (l < 16) ? 0 : 8;
  const float* n_c = wsF + WS_NC;
  _Float16* bp = (_Float16*)(wsF + WS_BPACK);
  v16h v;
  for (int h = 0; h < 8; ++h) {
    float x = (node < Ntot) ? n_c[(size_t)node * E_ + e0 + h] : 0.f;
    v[h] = (_Float16)x;
  }
  for (int h = 8; h < 16; ++h) v[h] = (_Float16)0.f;
  *(v16h*)(bp + (size_t)gid * 16) = v;
}

/* WMMA logits GEMM + fused online log-softmax.
   One wave32 per 16 target rows; B panels streamed into LDS by the Tensor Data
   Mover (double buffered), synchronized with s_wait_tensorcnt. */
__global__ __launch_bounds__(32)
void k_loss(const float* __restrict__ comp_l, const float* __restrict__ comp_r,
            const float* __restrict__ cbp, int* wsI, float* wsF) {
  __shared__ _Float16 lbuf[2][32 * 16];          /* 2 x 1KB B-panel slots */
  __shared__ float lse[16];
  const int lane = threadIdx.x;
  const int tile = blockIdx.x;
  const int Ntot = wsI[0];
  const int Tval = wsI[1];
  const float* n_c = wsF + WS_NC;
  const int* tc = wsI + WS_TCMP;
  float* lossv = wsF + WS_LOSS;
  const _Float16* bp = (const _Float16*)(wsF + WS_BPACK);

  float sl[4], sr[4], cbv[4];
  for (int i = 0; i < 4; ++i) {
    sl[i] = 1.f / (1.f + __expf(-comp_l[i]));
    sr[i] = 1.f / (1.f + __expf(-comp_r[i]));
    cbv[i] = cbp[i];
  }

  /* A tile: lane holds row (lane&15), K-slice e0..e0+7; rows = comp(left,right) */
  const int rA  = lane & 15;
  const int e0  = (lane < 16) ? 0 : 8;
  const int row = tile * 16 + rA;
  const bool rowValid = (row < Tval);
  int li = 0, ri = 0;
  if (rowValid) { li = tc[row * 4 + 1]; ri = tc[row * 4 + 2]; }
  v16h a;
  for (int h = 0; h < 8; ++h) {
    float p = 0.f;
    if (rowValid) {
      int e = e0 + h;
      p = sl[e & 3] * n_c[(size_t)li * E_ + e] +
          sr[e & 3] * n_c[(size_t)ri * E_ + e] + cbv[e & 3];
    }
    a[h] = (_Float16)p;
  }
  for (int h = 8; h < 16; ++h) a[h] = (_Float16)0.f;

  float m[8], s[8];
  for (int v = 0; v < 8; ++v) { m[v] = -INFINITY; s[v] = 0.f; }

  const int ntiles = (Ntot + 15) >> 4;
  __builtin_prefetch(bp, 0, 0);                  /* global_prefetch_b8 */

#if HAVE_TDM
  /* TDM issue: 1-D tile of 256 dwords (one packed B panel) -> LDS slot.
     D# per CDNA5 ISA: group0 {count=1, lds_addr, global_addr,
     type=2}; group1 {data_size=4B, tensor_dim0=256, tensor_dim1=1,
     tile_dim0=256, tile_dim1=1, dim0_stride=256}; groups 2/3 zero (<=2D).
     This toolchain exposes the 6-arg builtin: (g0, g1, g2, g3, g4, cpol). */
  auto tdm_issue = [&](int j, int slot) {
    /* flat shared-aperture addr: low 32 bits are the LDS byte offset */
    unsigned lds_off = (unsigned)(uintptr_t)&lbuf[slot][0];
    unsigned long long ga = (unsigned long long)(uintptr_t)(bp + (size_t)j * 512);
    u32x4 g0 = { 1u, lds_off, (unsigned)(ga & 0xffffffffu),
                 (unsigned)((ga >> 32) & 0x01ffffffu) | (2u << 30) };
    i32x8 g1 = { (int)(2u << 16),        /* data_size = 4B            */
                 (int)(256u << 16),      /* tensor_dim0.lo = 256      */
                 (int)(1u << 16),        /* tdim0.hi=0, tensor_dim1=1 */
                 (int)(256u << 16),      /* tdim1.hi=0, tile_dim0=256 */
                 1,                      /* tile_dim1 = 1             */
                 256,                    /* tensor_dim0_stride = 256  */
                 0, 0 };
    i32x4 gz4 = { 0, 0, 0, 0 };
    i32x8 gz8 = { 0, 0, 0, 0, 0, 0, 0, 0 };
    __builtin_amdgcn_tensor_load_to_lds(g0, g1, gz4, gz4, gz8, 0);
  };
#endif

  auto consume = [&](int j) {
#if HAVE_TDM
    v16h bb = *(const v16h*)&lbuf[j & 1][lane * 16];   /* ds_load_b128 x2 */
#else
    v16h bb = *(const v16h*)(bp + ((size_t)j * 32 + lane) * 16);
#endif
    v8f c = {};
    c = __builtin_amdgcn_wmma_f32_16x16x32_f16(false, a, false, bb,
                                               (short)0, c, false, false);
    const int node = j * 16 + (lane & 15);
    if (node < Ntot) {
      for (int v = 0; v < 8; ++v) {          /* per-lane online softmax */
        float x = c[v];
        float nm = fmaxf(m[v], x);
        s[v] = s[v] * __expf(m[v] - nm) + __expf(x - nm);
        m[v] = nm;
      }
    }
  };

#if HAVE_TDM
  tdm_issue(0, 0);
  int j = 0;
  for (; j + 1 < ntiles; ++j) {
    tdm_issue(j + 1, (j + 1) & 1);
    __builtin_amdgcn_s_wait_tensorcnt(1);    /* oldest transfer done */
    consume(j);
  }
  __builtin_amdgcn_s_wait_tensorcnt(0);
  consume(j);
#else
  for (int j = 0; j < ntiles; ++j) consume(j);
#endif

  /* merge the 16 column-lanes of each half-group */
  for (int off = 1; off < 16; off <<= 1) {
    for (int v = 0; v < 8; ++v) {
      float om = __shfl_xor(m[v], off, 32);
      float os = __shfl_xor(s[v], off, 32);
      float nm = fmaxf(m[v], om);
      s[v] = s[v] * __expf(m[v] - nm) + os * __expf(om - nm);
      m[v] = nm;
    }
  }
  if (lane == 0)  for (int v = 0; v < 8; ++v) lse[v]     = m[v] + __logf(s[v]);
  if (lane == 16) for (int v = 0; v < 8; ++v) lse[8 + v] = m[v] + __logf(s[v]);
  __syncthreads();

  if (lane < 16) {
    int r = tile * 16 + lane;
    if (r < Tval) {
      int lbl = tc[r * 4 + 0], l2 = tc[r * 4 + 1], r2 = tc[r * 4 + 2];
      float d = 0.f;
      for (int e = 0; e < E_; ++e) {
        float p = sl[e & 3] * n_c[(size_t)l2 * E_ + e] +
                  sr[e & 3] * n_c[(size_t)r2 * E_ + e] + cbv[e & 3];
        d += p * n_c[(size_t)lbl * E_ + e];
      }
      lossv[r] = lse[lane] - d;               /* NLL for this row */
    }
  }
}

__global__ void k_final(int* wsI, float* wsF, float* out) {
  if (threadIdx.x != 0 || blockIdx.x != 0) return;
  const int T = wsI[1];
  const float* lo = wsF + WS_LOSS;
  float s = 0.f;
  for (int i = 0; i < T; ++i) s += lo[i];     /* fixed order -> deterministic */
  out[0] = (T > 0) ? (s / (float)T) : 0.f;
}

extern "C" void kernel_launch(void* const* d_in, const int* in_sizes, int n_in,
                              void* d_out, int out_size, void* d_ws, size_t ws_size,
                              hipStream_t stream) {
  (void)in_sizes; (void)n_in; (void)out_size; (void)ws_size;
  const int*   seqs = (const int*)d_in[0];
  const float* emb  = (const float*)d_in[1];
  const float* cl   = (const float*)d_in[2];
  const float* cr   = (const float*)d_in[3];
  const float* cb   = (const float*)d_in[4];
  float* out = (float*)d_out;
  int*   wsI = (int*)d_ws;
  float* wsF = (float*)d_ws;

  hipLaunchKernelGGL(k_zero,    dim3((WS_TOTAL + 255) / 256), dim3(256), 0, stream, wsI);
  hipLaunchKernelGGL(k_mark,    dim3((B_ * S_ + 255) / 256),  dim3(256), 0, stream, seqs, wsI);
  hipLaunchKernelGGL(k_scan,    dim3(1),    dim3(1024), 0, stream, emb, wsI, wsF);
  hipLaunchKernelGGL(k_build,   dim3(B_),   dim3(32),   0, stream, seqs, cl, cr, cb, wsI, wsF);
  hipLaunchKernelGGL(k_compact, dim3(1),    dim3(1),    0, stream, wsI);
  hipLaunchKernelGGL(k_pack,    dim3((NTCAP * 32 + 255) / 256), dim3(256), 0, stream, wsI, wsF);
  hipLaunchKernelGGL(k_loss,    dim3(TILES),dim3(32),   0, stream, cl, cr, cb, wsI, wsF);
  hipLaunchKernelGGL(k_final,   dim3(1),    dim3(1),    0, stream, wsI, wsF, out);
}
